// Attention_70901320122819
// MI455X (gfx1250) — matrix operand order repaired and checked by reading, not compiled, not run on hardware
//
#include <hip/hip_runtime.h>
#include <hip/hip_bf16.h>

// ---------------------------------------------------------------------------
// Fused QKV projection + causal multi-head attention for gfx1250 (CDNA5).
//   x      : [4, 2048, 1024] f32
//   W_qkv  : [1024, 3072]    f32
//   b_qkv  : [3072]          f32
//   out    : [4, 2048, 1024] f32
// Pipeline: (1) one-shot fp32->bf16 conversion of x and W^T (memory-bound,
// removes all conversion VALU from the GEMM hot loop), (2) WMMA QKV GEMM,
// (3) WMMA causal flash attention. All fragments are b128 loads in contiguous k order
// (k = 16*half + s, same for A and B); matrix math is v_wmma_f32_16x16x32_bf16 (wave32).
// ---------------------------------------------------------------------------

typedef __attribute__((ext_vector_type(16))) __bf16          v16bf;
typedef __attribute__((ext_vector_type(16))) unsigned short  v16u;
typedef __attribute__((ext_vector_type(8)))  float           v8f;
typedef __attribute__((ext_vector_type(4)))  unsigned int    v4u;
typedef __attribute__((ext_vector_type(2)))  unsigned int    v2u;
typedef __attribute__((ext_vector_type(4)))  float           v4f;

#if defined(__has_builtin)
#if __has_builtin(__builtin_amdgcn_cvt_pk_bf16_f32)
#define HAVE_CVT_PK_BF16 1
#endif
#endif
#ifndef HAVE_CVT_PK_BF16
#define HAVE_CVT_PK_BF16 0
#endif

union Frag16 {
    v16u  u;
    v16bf b;
    v4u   q[2];
};

#define BATCH 4
#define SEQ   2048
#define DIMD  1024
#define HEADS 16
#define HDIM  64
#define N3    3072
#define MROWS 8192   // BATCH*SEQ

// packed fp32x2 -> bf16x2 (hardware cvt when available, manual RNE otherwise)
__device__ __forceinline__ unsigned pk2bf(float lo, float hi) {
#if HAVE_CVT_PK_BF16
    auto t = __builtin_amdgcn_cvt_pk_bf16_f32(lo, hi);
    return __builtin_bit_cast(unsigned, t);
#else
    unsigned ul = __float_as_uint(lo), uh = __float_as_uint(hi);
    ul = (ul + 0x7FFFu + ((ul >> 16) & 1u)) >> 16;
    uh = (uh + 0x7FFFu + ((uh >> 16) & 1u)) & 0xFFFF0000u;
    return ul | uh;
#endif
}
__device__ __forceinline__ unsigned short f2bf(float f) {
    return (unsigned short)pk2bf(f, 0.0f);
}

__device__ __forceinline__ v8f wmma_bf16(const Frag16& a, const Frag16& b, v8f c) {
    return __builtin_amdgcn_wmma_f32_16x16x32_bf16(
        /*neg_a=*/false, a.b, /*neg_b=*/false, b.b,
        /*c_mod=*/(short)0, c, /*reuse_a=*/false, /*reuse_b=*/false);
}

__device__ __forceinline__ v4u ld128(const unsigned short* p) {
    return *reinterpret_cast<const v4u*>(p);
}

// ---------------------------------------------------------------------------
// Prep 1: x fp32 -> bf16, contiguous. One v4f load + one v2u store per thread.
// ---------------------------------------------------------------------------
__global__ __launch_bounds__(256)
void convert_x_bf16(const float* __restrict__ x, unsigned short* __restrict__ xb)
{
    const size_t i = ((size_t)blockIdx.x * 256 + threadIdx.x) * 4;
    const v4f f = *reinterpret_cast<const v4f*>(x + i);
    v2u pk;
    pk.x = pk2bf(f.x, f.y);
    pk.y = pk2bf(f.z, f.w);
    *reinterpret_cast<v2u*>(xb + i) = pk;
}

// ---------------------------------------------------------------------------
// Prep 2: W [1024][3072] fp32 -> W^T [3072][1024] bf16. Writes are contiguous
// b128; strided reads are absorbed by the 192MB L2 (W is only 12MB).
// ---------------------------------------------------------------------------
__global__ __launch_bounds__(256)
void convert_wT_bf16(const float* __restrict__ W, unsigned short* __restrict__ wt)
{
    const int idx = blockIdx.x * 256 + threadIdx.x;   // 393216 threads
    const int n  = idx >> 7;                          // output row (0..3071)
    const int k0 = (idx & 127) * 8;                   // k chunk (0..1016)
    const float* src = W + (size_t)k0 * N3 + n;
    v4u pk;
#pragma unroll
    for (int j = 0; j < 4; ++j)
        pk[j] = pk2bf(src[(size_t)(2 * j) * N3], src[(size_t)(2 * j + 1) * N3]);
    *reinterpret_cast<v4u*>(wt + (size_t)n * DIMD + k0) = pk;
}

// ---------------------------------------------------------------------------
// Kernel 3: qkv = x @ W + b (bf16 inputs from ws). q is pre-scaled by
// 1/sqrt(1024) so the attention kernel skips logit scaling. q/k stored
// [b*h][t][64]; v stored TRANSPOSED [b*h][64][t].
// Block: 256 threads = 8 waves; block tile 64(M) x 128(N); K-step 32.
// Hot loop has no conversion math: b128 global->LDS copies + b128 fragment
// loads + 4 WMMAs.
// ---------------------------------------------------------------------------
__global__ __launch_bounds__(256)
void qkv_gemm_wmma(const unsigned short* __restrict__ xb,
                   const unsigned short* __restrict__ wt,
                   const float* __restrict__ bias,
                   unsigned short* __restrict__ q_ws,
                   unsigned short* __restrict__ k_ws,
                   unsigned short* __restrict__ v_ws)
{
    __shared__ unsigned short lds_a[64 * 32];    // A tile, row-major [m][k]
    __shared__ unsigned short lds_bT[128 * 32];  // B tile, col-major [n][k]

    const int tid  = threadIdx.x;
    const int lane = tid & 31;
    const int wid  = tid >> 5;
    const int hi   = lane >> 4;      // half-wave select (wave32 WMMA layout)
    const int ln   = lane & 15;

    const int m0 = blockIdx.y * 64;
    const int n0 = blockIdx.x * 128;
    const int wm = (wid & 3) * 16;   // wave row inside block tile
    const int wn = (wid >> 2) * 64;  // wave col inside block tile

    // ---- staging pointers (bumped by 32 each K-step) ----
    const int eA = tid * 8;                        // A: 8 bf16 per thread
    const unsigned short* pA = xb + (size_t)(m0 + (eA >> 5)) * DIMD + (eA & 31);
    const int nB = tid >> 1;                       // B: 16 bf16 per thread
    const int kB = (tid & 1) * 16;
    const int eB = nB * 32 + kB;
    const unsigned short* pB = wt + (size_t)(n0 + nB) * DIMD + kB;

    v8f acc[4];
#pragma unroll
    for (int i = 0; i < 4; ++i)
#pragma unroll
        for (int e = 0; e < 8; ++e) acc[i][e] = 0.0f;

    for (int kk = 0; kk < DIMD; kk += 32) {
        // ---- stage: pure b128 copies, no conversion ----
        *reinterpret_cast<v4u*>(&lds_a[eA])      = ld128(pA);
        *reinterpret_cast<v4u*>(&lds_bT[eB])     = ld128(pB);
        *reinterpret_cast<v4u*>(&lds_bT[eB + 8]) = ld128(pB + 8);
        pA += 32;
        pB += 32;
        if (kk + 32 < DIMD)
            __builtin_prefetch(pA, 0, 1);
        __syncthreads();

        // ---- A fragment: two contiguous 16B chunks of one row ----
        Frag16 fa;
        {
            const unsigned short* ap = &lds_a[(wm + ln) * 32 + hi * 16];
            fa.q[0] = ld128(ap);            // kloc = hi*16 + 0..7
            fa.q[1] = ld128(ap + 8);        // kloc = hi*16 + 8..15
        }
        // ---- 4 B fragments + WMMA ----
#pragma unroll
        for (int nt = 0; nt < 4; ++nt) {
            Frag16 fb;
            const unsigned short* bp = &lds_bT[(wn + nt * 16 + ln) * 32 + hi * 16];
            fb.q[0] = ld128(bp);            // kloc = hi*16 + 0..7
            fb.q[1] = ld128(bp + 8);        // kloc = hi*16 + 8..15
            acc[nt] = wmma_bf16(fa, fb, acc[nt]);
        }
        __syncthreads();
    }

    // ---- epilogue: bias add; all selector math is wave-uniform ----
    const int nbase  = n0 + wn;           // 64-aligned => one head, one third
    const int which  = nbase >> 10;       // 0=q 1=k 2=v
    const int hconst = (nbase & 1023) >> 6;
    const int mbase  = m0 + wm;           // 16-aligned inside one batch block
    const int bb     = mbase >> 11;
    const int tbase  = (mbase & 2047) + 8 * hi;
    const size_t bhc = (size_t)(bb * HEADS + hconst);

    if (which == 2) {
        // v transposed [b*h][hd][t]: 8 consecutive t per lane -> one b128 store
#pragma unroll
        for (int nt = 0; nt < 4; ++nt) {
            const int hd = nt * 16 + ln;
            const float bv = bias[nbase + nt * 16 + ln];
            unsigned short* col = v_ws + (bhc * HDIM + hd) * SEQ + tbase;
            v4u pk;
#pragma unroll
            for (int e = 0; e < 4; ++e)
                pk[e] = pk2bf(acc[nt][2 * e] + bv, acc[nt][2 * e + 1] + bv);
            *reinterpret_cast<v4u*>(col) = pk;
        }
    } else {
        unsigned short* dst = (which == 0) ? q_ws : k_ws;   // [b*h][t][64]
        const float scale = (which == 0) ? 0.03125f : 1.0f; // fold 1/sqrt(d) into q
#pragma unroll
        for (int nt = 0; nt < 4; ++nt) {
            const int hd = nt * 16 + ln;
            const float bv = bias[nbase + nt * 16 + ln];
#pragma unroll
            for (int r = 0; r < 8; ++r)
                dst[(bhc * SEQ + (tbase + r)) * HDIM + hd] =
                    f2bf((acc[nt][r] + bv) * scale);
        }
    }
}

// ---------------------------------------------------------------------------
// Kernel 4: causal flash attention. One wave owns 16 query rows; 8 waves per
// block. Per 32-key chunk: 4 WMMAs for S=Q*K^T (b128 K loads), online softmax
// (shfl_xor within 16-lane half-groups), P permuted via per-wave LDS
// (scalar scatter, b128 gather), 4 WMMAs for P*V (b128 loads from V^T).
// No block barriers: key-loop trip count differs per wave.
// ---------------------------------------------------------------------------
__global__ __launch_bounds__(256)
void flash_attn_wmma(const unsigned short* __restrict__ q_ws,
                     const unsigned short* __restrict__ k_ws,
                     const unsigned short* __restrict__ v_ws,
                     float* __restrict__ out)
{
    __shared__ unsigned short lds_p[8][16 * 32];   // per-wave P staging

    const int tid  = threadIdx.x;
    const int lane = tid & 31;
    const int wid  = tid >> 5;
    const int hi   = lane >> 4;
    const int ln   = lane & 15;

    const int bh    = blockIdx.y;                  // b*16 + h
    const int q0    = blockIdx.x * 128 + wid * 16; // first query row of wave
    const int batch = bh >> 4;
    const int head  = bh & 15;

    const unsigned short* qb = q_ws + (size_t)bh * SEQ * HDIM;
    const unsigned short* kp = k_ws + (size_t)bh * SEQ * HDIM;
    const unsigned short* vp = v_ws + (size_t)bh * HDIM * SEQ;  // [hd][t]

    // Q (pre-scaled by 1/32): 16 rows x 64 hd -> two A fragments
    Frag16 aq[2];
    {
        const unsigned short* qrow = qb + (size_t)(q0 + ln) * HDIM;
#pragma unroll
        for (int h2 = 0; h2 < 2; ++h2) {
            aq[h2].q[0] = ld128(qrow + h2 * 32 + hi * 16);
            aq[h2].q[1] = ld128(qrow + h2 * 32 + 8 + hi * 16);
        }
    }

    // hoisted per-lane base pointers (in-loop address = base + kb*stride)
    const unsigned short* kbase0 = kp + (size_t)ln * HDIM + hi * 16;
    const unsigned short* kbase1 = kp + (size_t)(16 + ln) * HDIM + hi * 16;
    const unsigned short* vbase[4];
#pragma unroll
    for (int nt = 0; nt < 4; ++nt)
        vbase[nt] = vp + (size_t)(nt * 16 + ln) * SEQ + hi * 16;

    float m_i[8], l_i[8];
    v8f accv[4];
#pragma unroll
    for (int r = 0; r < 8; ++r) { m_i[r] = -__builtin_inff(); l_i[r] = 0.0f; }
#pragma unroll
    for (int nt = 0; nt < 4; ++nt)
#pragma unroll
        for (int e = 0; e < 8; ++e) accv[nt][e] = 0.0f;

    const int qmax = q0 + 15;
    for (int kb = 0; kb <= qmax; kb += 32) {   // kb multiple of 32 => keys <= 2047
        // ---- S = Q * K^T for two 16-key subtiles ----
        v8f s[2];
#pragma unroll
        for (int t = 0; t < 2; ++t) {
            const unsigned short* krow =
                (t == 0 ? kbase0 : kbase1) + (size_t)kb * HDIM;
            Frag16 fb0, fb1;
            fb0.q[0] = ld128(krow);            // hd  0..31 (this half-wave)
            fb0.q[1] = ld128(krow + 8);
            fb1.q[0] = ld128(krow + 32);       // hd 32..63
            fb1.q[1] = ld128(krow + 40);
            v8f z;
#pragma unroll
            for (int e = 0; e < 8; ++e) z[e] = 0.0f;
            z = wmma_bf16(aq[0], fb0, z);
            z = wmma_bf16(aq[1], fb1, z);
            s[t] = z;
        }

        // ---- causal mask + online softmax (logits already scaled) ----
#pragma unroll
        for (int r = 0; r < 8; ++r) {
            const int qr = q0 + r + 8 * hi;
            float a0 = s[0][r];
            float a1 = s[1][r];
            if (kb + ln      > qr) a0 = -__builtin_inff();
            if (kb + 16 + ln > qr) a1 = -__builtin_inff();
            float mx = fmaxf(a0, a1);
            mx = fmaxf(mx, __shfl_xor(mx, 1));
            mx = fmaxf(mx, __shfl_xor(mx, 2));
            mx = fmaxf(mx, __shfl_xor(mx, 4));
            mx = fmaxf(mx, __shfl_xor(mx, 8));
            const float mnew  = fmaxf(m_i[r], mx);
            const float alpha = __expf(m_i[r] - mnew);
            m_i[r] = mnew;
            const float p0 = __expf(a0 - mnew);
            const float p1 = __expf(a1 - mnew);
            s[0][r] = p0;
            s[1][r] = p1;
            float rs = p0 + p1;
            rs += __shfl_xor(rs, 1);
            rs += __shfl_xor(rs, 2);
            rs += __shfl_xor(rs, 4);
            rs += __shfl_xor(rs, 8);
            l_i[r] = l_i[r] * alpha + rs;
#pragma unroll
            for (int nt = 0; nt < 4; ++nt) accv[nt][r] *= alpha;
        }

        // ---- P: C-layout -> LDS scatter -> A-layout b128 gather ----
#pragma unroll
        for (int t = 0; t < 2; ++t)
#pragma unroll
            for (int r = 0; r < 8; ++r)
                lds_p[wid][(r + 8 * hi) * 32 + t * 16 + ln] = f2bf(s[t][r]);

        Frag16 pa;
        {
            const unsigned short* pp = &lds_p[wid][ln * 32 + hi * 16];
            pa.q[0] = ld128(pp);
            pa.q[1] = ld128(pp + 8);
        }

        // ---- O += P * V (V stored transposed: key index contiguous) ----
#pragma unroll
        for (int nt = 0; nt < 4; ++nt) {
            Frag16 fv;
            const unsigned short* vrow = vbase[nt] + kb;
            fv.q[0] = ld128(vrow);            // keys kb + hi*16 + 0..7
            fv.q[1] = ld128(vrow + 8);        // keys kb + hi*16 + 8..15
            accv[nt] = wmma_bf16(pa, fv, accv[nt]);
        }
    }

    // ---- epilogue: normalize (reciprocal) and write [b, t, d] fp32 ----
    float rinv[8];
#pragma unroll
    for (int r = 0; r < 8; ++r) rinv[r] = 1.0f / l_i[r];

    float* orow = out + ((size_t)(batch * SEQ + q0 + 8 * hi)) * DIMD + head * HDIM;
#pragma unroll
    for (int nt = 0; nt < 4; ++nt) {
        const int col = nt * 16 + ln;
#pragma unroll
        for (int r = 0; r < 8; ++r)
            orow[(size_t)r * DIMD + col] = accv[nt][r] * rinv[r];
    }
}

// ---------------------------------------------------------------------------
extern "C" void kernel_launch(void* const* d_in, const int* in_sizes, int n_in,
                              void* d_out, int out_size, void* d_ws, size_t ws_size,
                              hipStream_t stream)
{
    const float* x    = (const float*)d_in[0];   // [4,2048,1024]
    const float* W    = (const float*)d_in[1];   // [1024,3072]
    const float* bias = (const float*)d_in[2];   // [3072]
    float* out        = (float*)d_out;

    // workspace layout (ushort elements)
    const size_t n_x  = (size_t)MROWS * DIMD;    // 8,388,608
    const size_t n_w  = (size_t)DIMD * N3;       // 3,145,728
    const size_t n_t  = (size_t)BATCH * HEADS * SEQ * HDIM;  // 8,388,608
    unsigned short* xb   = (unsigned short*)d_ws;
    unsigned short* wt   = xb + n_x;
    unsigned short* q_ws = wt + n_w;
    unsigned short* k_ws = q_ws + n_t;
    unsigned short* v_ws = k_ws + n_t;

    convert_x_bf16 <<<dim3((unsigned)(n_x / 4 / 256)), 256, 0, stream>>>(x, xb);
    convert_wT_bf16<<<dim3((unsigned)(n_w / 8 / 256)), 256, 0, stream>>>(W, wt);

    dim3 g1(N3 / 128, MROWS / 64);     // 24 x 128 blocks
    qkv_gemm_wmma<<<g1, 256, 0, stream>>>(xb, wt, bias, q_ws, k_ws, v_ws);

    dim3 g2(SEQ / 128, BATCH * HEADS); // 16 x 64 blocks
    flash_attn_wmma<<<g2, 256, 0, stream>>>(q_ws, k_ws, v_ws, out);
}
